// GRANMixtureBernoulli_86002425135277
// MI455X (gfx1250) — matrix operand order repaired and checked by reading, not compiled, not run on hardware
//
#include <hip/hip_runtime.h>
#include <hip/hip_bf16.h>

// ---------------------------------------------------------------------------
// GRAN mixture-Bernoulli forward for MI455X (gfx1250, wave32, WMMA).
// Heavy path: 16 batched 256x256x256 GEMMs -> V_WMMA_F32_16X16X32_BF16.
// Block = 8 waves computing a 64x128 macro-tile; each wave owns a 32x32
// register tile (2x2 WMMA accumulators). Operands staged through
// double-buffered LDS (padded rows, 16B-aligned b128 chunks) so L2 traffic
// is ~44 FLOP/byte instead of ~8 with per-wave direct loads.
// ---------------------------------------------------------------------------

typedef __bf16 bf16;
typedef __attribute__((ext_vector_type(8)))  bf16  v8bf;
typedef __attribute__((ext_vector_type(16))) bf16  v16bf;
typedef __attribute__((ext_vector_type(8)))  float v8f;

#define B_   128
#define N_   256
#define L_   4
#define K_   10
#define NLg  16   // NL == NG == 16

#define SHUF16(lo, hi) __builtin_shufflevector(lo, hi, 0,1,2,3,4,5,6,7,8,9,10,11,12,13,14,15)
#define WMMA_BF16(a, b, c) __builtin_amdgcn_wmma_f32_16x16x32_bf16(false, a, false, b, (short)0, c, false, false)

// ---------------------------------------------------------------------------
// small utility kernels
// ---------------------------------------------------------------------------
__global__ void k_zero(float* p, int n) {
  int i = blockIdx.x * blockDim.x + threadIdx.x;
  if (i < n) p[i] = 0.f;
}

__global__ void k_f2bf(const float* __restrict__ s, bf16* __restrict__ d, int n) {
  int i = blockIdx.x * blockDim.x + threadIdx.x;
  if (i < n) d[i] = (bf16)s[i];
}

// WT[l][n][k] = W[l][k][n]  (column-major weights -> contiguous K per lane)
__global__ void k_wt(const float* __restrict__ W, bf16* __restrict__ WT, int layers) {
  int i = blockIdx.x * blockDim.x + threadIdx.x;
  int tot = layers * N_ * N_;
  if (i >= tot) return;
  int l = i / (N_ * N_), r = i % (N_ * N_);
  int n = r / N_, k = r % N_;
  WT[i] = (bf16)W[(size_t)l * N_ * N_ + (size_t)k * N_ + n];
}

// X0 = identity (f32 current + bf16 transposed)
__global__ void k_ident(float* __restrict__ Xcur, bf16* __restrict__ XT) {
  int i = blockIdx.x * blockDim.x + threadIdx.x;
  if (i >= B_ * N_ * N_) return;
  int rc = i % (N_ * N_);
  float v = ((rc / N_) == (rc % N_)) ? 1.f : 0.f;
  Xcur[i] = v;
  XT[i] = (bf16)v;
}

// ---------------------------------------------------------------------------
// GEMM 1:  tmp = (1+eps)*X + A @ X      (bf16 out, row-major, f32 accum)
//   A-frag: adjacency rows (bf16 row-major)      [16x32 A layout]
//   B-frag: XT rows = X columns (contig bf16)    [32x16 B layout]
//   C-init: (1+eps) * Xcur (f32)
// ---------------------------------------------------------------------------
__global__ __launch_bounds__(256) void k_gemm1(
    const bf16* __restrict__ Abf, const bf16* __restrict__ XT,
    const float* __restrict__ Xcur, const float* __restrict__ eps, int eidx,
    bf16* __restrict__ tmp)
{
  __shared__ bf16 As[2][64][48];    // 64x32 tile, rows padded to 96B
  __shared__ bf16 Bs[2][128][48];   // 128x32 tile
  const int tid = threadIdx.x, wave = tid >> 5, lane = tid & 31;
  const int half = lane >> 4, r = lane & 15;
  const int bid = blockIdx.x;
  const int b = bid >> 3;
  const int mBlk = ((bid >> 1) & 3) * 64;
  const int nBlk = (bid & 1) * 128;
  const int wm = (wave >> 2) * 32;  // 0 / 32
  const int wn = (wave & 3) * 32;   // 0 / 32 / 64 / 96
  const float scale = 1.f + eps[eidx];

  // staging assignments: A (64 rows x 4 chunks), B (128 rows x 4 chunks, 2/thread)
  const int arow = tid >> 2, achk = (tid & 3) * 8;
  const int brow0 = (tid * 2) >> 2,     bchk0 = ((tid * 2) & 3) * 8;
  const int brow1 = (tid * 2 + 1) >> 2, bchk1 = ((tid * 2 + 1) & 3) * 8;
  const bf16* gA  = Abf + ((size_t)b * N_ + mBlk + arow) * N_ + achk;
  const bf16* gB0 = XT  + ((size_t)b * N_ + nBlk + brow0) * N_ + bchk0;
  const bf16* gB1 = XT  + ((size_t)b * N_ + nBlk + brow1) * N_ + bchk1;

  // C-init = (1+eps) * X
  v8f c[2][2];
#pragma unroll
  for (int mi = 0; mi < 2; ++mi)
#pragma unroll
    for (int ni = 0; ni < 2; ++ni) {
      const float* xp = Xcur + ((size_t)b * N_ + mBlk + wm + mi * 16 + half * 8) * N_
                        + nBlk + wn + ni * 16 + r;
#pragma unroll
      for (int i = 0; i < 8; ++i) c[mi][ni][i] = scale * xp[(size_t)i * N_];
    }

  // preload K-step 0
  v8bf ra  = *(const v8bf*)(gA);
  v8bf rb0 = *(const v8bf*)(gB0);
  v8bf rb1 = *(const v8bf*)(gB1);
  *(v8bf*)&As[0][arow][achk]   = ra;
  *(v8bf*)&Bs[0][brow0][bchk0] = rb0;
  *(v8bf*)&Bs[0][brow1][bchk1] = rb1;

  for (int s = 0; s < 8; ++s) {
    const int cur = s & 1, nxt = cur ^ 1;
    if (s < 7) {                     // prefetch next K-step into registers
      const int kk = (s + 1) * 32;
      ra  = *(const v8bf*)(gA + kk);
      rb0 = *(const v8bf*)(gB0 + kk);
      rb1 = *(const v8bf*)(gB1 + kk);
    }
    __syncthreads();                 // buf[cur] stores visible; buf[nxt] reads done

    v16bf a0, a1, bf0, bf1;
    {
      const bf16* p = &As[cur][wm + r][half * 8];
      a0 = SHUF16(*(const v8bf*)p, *(const v8bf*)(p + 16));
      p = &As[cur][wm + 16 + r][half * 8];
      a1 = SHUF16(*(const v8bf*)p, *(const v8bf*)(p + 16));
      p = &Bs[cur][wn + r][half * 8];
      bf0 = SHUF16(*(const v8bf*)p, *(const v8bf*)(p + 16));
      p = &Bs[cur][wn + 16 + r][half * 8];
      bf1 = SHUF16(*(const v8bf*)p, *(const v8bf*)(p + 16));
    }
    c[0][0] = WMMA_BF16(a0, bf0, c[0][0]);
    c[0][1] = WMMA_BF16(a0, bf1, c[0][1]);
    c[1][0] = WMMA_BF16(a1, bf0, c[1][0]);
    c[1][1] = WMMA_BF16(a1, bf1, c[1][1]);

    if (s < 7) {                     // drain prefetch regs into other buffer
      *(v8bf*)&As[nxt][arow][achk]   = ra;
      *(v8bf*)&Bs[nxt][brow0][bchk0] = rb0;
      *(v8bf*)&Bs[nxt][brow1][bchk1] = rb1;
    }
  }

  // epilogue: bf16 row-major tmp
#pragma unroll
  for (int mi = 0; mi < 2; ++mi)
#pragma unroll
    for (int ni = 0; ni < 2; ++ni) {
      bf16* out = tmp + ((size_t)b * N_ + mBlk + wm + mi * 16 + half * 8) * N_
                  + nBlk + wn + ni * 16 + r;
#pragma unroll
      for (int i = 0; i < 8; ++i) out[(size_t)i * N_] = (bf16)c[mi][ni][i];
    }
}

// ---------------------------------------------------------------------------
// GEMM 2:  Xnext = ReLU(tmp @ W + bias)
//   writes: Xcur (f32), optional z_l slice (f32), XT (bf16 transposed)
// ---------------------------------------------------------------------------
__global__ __launch_bounds__(256) void k_gemm2(
    const bf16* __restrict__ tmp, const bf16* __restrict__ WT,
    const float* __restrict__ bias,
    float* __restrict__ Xcur, float* __restrict__ zl, int zstride, int zoff,
    bf16* __restrict__ XT)
{
  __shared__ bf16 As[2][64][48];
  __shared__ bf16 Bs[2][128][48];
  const int tid = threadIdx.x, wave = tid >> 5, lane = tid & 31;
  const int half = lane >> 4, r = lane & 15;
  const int bid = blockIdx.x;
  const int b = bid >> 3;
  const int mBlk = ((bid >> 1) & 3) * 64;
  const int nBlk = (bid & 1) * 128;
  const int wm = (wave >> 2) * 32;
  const int wn = (wave & 3) * 32;

  const int arow = tid >> 2, achk = (tid & 3) * 8;
  const int brow0 = (tid * 2) >> 2,     bchk0 = ((tid * 2) & 3) * 8;
  const int brow1 = (tid * 2 + 1) >> 2, bchk1 = ((tid * 2 + 1) & 3) * 8;
  const bf16* gA  = tmp + ((size_t)b * N_ + mBlk + arow) * N_ + achk;
  const bf16* gB0 = WT + (size_t)(nBlk + brow0) * N_ + bchk0;   // weights: no batch dim
  const bf16* gB1 = WT + (size_t)(nBlk + brow1) * N_ + bchk1;

  v8f c[2][2];
#pragma unroll
  for (int mi = 0; mi < 2; ++mi)
#pragma unroll
    for (int ni = 0; ni < 2; ++ni) {
      const float bv0 = bias[nBlk + wn + ni * 16 + r];
#pragma unroll
      for (int i = 0; i < 8; ++i) c[mi][ni][i] = bv0;
    }

  v8bf ra  = *(const v8bf*)(gA);
  v8bf rb0 = *(const v8bf*)(gB0);
  v8bf rb1 = *(const v8bf*)(gB1);
  *(v8bf*)&As[0][arow][achk]   = ra;
  *(v8bf*)&Bs[0][brow0][bchk0] = rb0;
  *(v8bf*)&Bs[0][brow1][bchk1] = rb1;

  for (int s = 0; s < 8; ++s) {
    const int cur = s & 1, nxt = cur ^ 1;
    if (s < 7) {
      const int kk = (s + 1) * 32;
      ra  = *(const v8bf*)(gA + kk);
      rb0 = *(const v8bf*)(gB0 + kk);
      rb1 = *(const v8bf*)(gB1 + kk);
    }
    __syncthreads();

    v16bf a0, a1, bf0, bf1;
    {
      const bf16* p = &As[cur][wm + r][half * 8];
      a0 = SHUF16(*(const v8bf*)p, *(const v8bf*)(p + 16));
      p = &As[cur][wm + 16 + r][half * 8];
      a1 = SHUF16(*(const v8bf*)p, *(const v8bf*)(p + 16));
      p = &Bs[cur][wn + r][half * 8];
      bf0 = SHUF16(*(const v8bf*)p, *(const v8bf*)(p + 16));
      p = &Bs[cur][wn + 16 + r][half * 8];
      bf1 = SHUF16(*(const v8bf*)p, *(const v8bf*)(p + 16));
    }
    c[0][0] = WMMA_BF16(a0, bf0, c[0][0]);
    c[0][1] = WMMA_BF16(a0, bf1, c[0][1]);
    c[1][0] = WMMA_BF16(a1, bf0, c[1][0]);
    c[1][1] = WMMA_BF16(a1, bf1, c[1][1]);

    if (s < 7) {
      *(v8bf*)&As[nxt][arow][achk]   = ra;
      *(v8bf*)&Bs[nxt][brow0][bchk0] = rb0;
      *(v8bf*)&Bs[nxt][brow1][bchk1] = rb1;
    }
  }

#pragma unroll
  for (int mi = 0; mi < 2; ++mi)
#pragma unroll
    for (int ni = 0; ni < 2; ++ni) {
#pragma unroll
      for (int i = 0; i < 8; ++i) {
        const int m = mBlk + wm + mi * 16 + half * 8 + i;
        const int n = nBlk + wn + ni * 16 + r;
        float v = fmaxf(c[mi][ni][i], 0.f);
        Xcur[((size_t)b * N_ + m) * N_ + n] = v;
        if (zl) zl[((size_t)b * N_ + m) * zstride + zoff + n] = v;
        XT[((size_t)b * N_ + n) * N_ + m] = (bf16)v;
      }
    }
}

// z_g[b][l*N + c] = sum_m X[b][m][c]
__global__ void k_colsum(const float* __restrict__ X, float* __restrict__ zg, int l) {
  int b = blockIdx.x, c = threadIdx.x;
  float s = 0.f;
  for (int m = 0; m < N_; ++m) s += X[((size_t)b * N_ + m) * N_ + c];
  zg[(size_t)b * (N_ * L_) + l * N_ + c] = s;
}

// ---------------------------------------------------------------------------
// clustering path
// ---------------------------------------------------------------------------
__global__ void k_cluster_mlp(const float* __restrict__ zl, const float* __restrict__ W,
                              const float* __restrict__ bias, float* __restrict__ h) {
  int i = blockIdx.x * blockDim.x + threadIdx.x;  // B*N*K
  if (i >= B_ * N_ * K_) return;
  int k = i % K_, bn = i / K_;
  const float* x = zl + (size_t)bn * (N_ * L_);
  float s = bias[k];
  for (int j = 0; j < N_ * L_; ++j) s += x[j] * W[j * K_ + k];
  h[i] = fmaxf(s, 0.f);
}

// per-node-channel BN stats over (B,K): stats[2n]=mean, stats[2n+1]=invstd
__global__ void k_cl_stats(const float* __restrict__ h, float* __restrict__ stats) {
  int n = blockIdx.x, t = threadIdx.x;
  __shared__ float s1[256], s2[256];
  float a = 0.f, q = 0.f;
  for (int i = t; i < B_ * K_; i += 256) {
    int b = i / K_, k = i % K_;
    float v = h[((size_t)b * N_ + n) * K_ + k];
    a += v; q += v * v;
  }
  s1[t] = a; s2[t] = q; __syncthreads();
  for (int st = 128; st > 0; st >>= 1) {
    if (t < st) { s1[t] += s1[t + st]; s2[t] += s2[t + st]; }
    __syncthreads();
  }
  if (t == 0) {
    float m = s1[0] / (float)(B_ * K_);
    float v = s2[0] / (float)(B_ * K_) - m * m;
    stats[2 * n] = m; stats[2 * n + 1] = rsqrtf(v + 1e-5f);
  }
}

// BN affine + softmax over node dim, per (b,k); in-place on h
__global__ void k_cl_softmax(float* __restrict__ h, const float* __restrict__ stats,
                             const float* __restrict__ g, const float* __restrict__ be) {
  int bk = blockIdx.x, b = bk / K_, k = bk % K_;
  int n = threadIdx.x;
  __shared__ float red[256];
  float v = (h[((size_t)b * N_ + n) * K_ + k] - stats[2 * n]) * stats[2 * n + 1] * g[n] + be[n];
  red[n] = v; __syncthreads();
  for (int st = 128; st > 0; st >>= 1) {
    if (n < st) red[n] = fmaxf(red[n], red[n + st]);
    __syncthreads();
  }
  float mx = red[0]; __syncthreads();
  float e = __expf(v - mx);
  red[n] = e; __syncthreads();
  for (int st = 128; st > 0; st >>= 1) {
    if (n < st) red[n] += red[n + st];
    __syncthreads();
  }
  h[((size_t)b * N_ + n) * K_ + k] = e / red[0];
}

__global__ void k_argmax(const float* __restrict__ h, int* __restrict__ idx) {
  int i = blockIdx.x * blockDim.x + threadIdx.x;  // B*N
  if (i >= B_ * N_) return;
  const float* p = h + (size_t)i * K_;
  int best = 0; float bvv = p[0];
  for (int k = 1; k < K_; ++k) if (p[k] > bvv) { bvv = p[k]; best = k; }
  idx[i] = best;
}

// z_l_c[b][k*1024 + f] = mean over nodes assigned to cluster k of z_l[b][n][f]
__global__ void k_zlc(const float* __restrict__ zl, const int* __restrict__ idx,
                      float* __restrict__ zlc) {
  int b = blockIdx.x / K_, k = blockIdx.x % K_;
  int t = threadIdx.x;  // 256, each owns 4 features
  float acc0 = 0.f, acc1 = 0.f, acc2 = 0.f, acc3 = 0.f;
  int cnt = 0;
  const int* ib = idx + b * N_;
  for (int n = 0; n < N_; ++n) {
    if (ib[n] == k) {
      ++cnt;
      const float* row = zl + ((size_t)b * N_ + n) * (N_ * L_);
      acc0 += row[t]; acc1 += row[t + 256]; acc2 += row[t + 512]; acc3 += row[t + 768];
    }
  }
  float inv = cnt > 0 ? 1.f / (float)cnt : 0.f;
  float* o = zlc + (size_t)b * (K_ * N_ * L_) + (size_t)k * (N_ * L_);
  o[t] = acc0 * inv; o[t + 256] = acc1 * inv; o[t + 512] = acc2 * inv; o[t + 768] = acc3 * inv;
}

// ---------------------------------------------------------------------------
// generic small dense + BN + elementwise
// ---------------------------------------------------------------------------
// act: 0=none, 1=relu, 2=sigmoid
__global__ void k_linear(const float* __restrict__ x, const float* __restrict__ W,
                         const float* __restrict__ bias, float* __restrict__ y,
                         int Bn, int In, int On, int act) {
  int i = blockIdx.x * blockDim.x + threadIdx.x;
  if (i >= Bn * On) return;
  int o = i % On, bb = i / On;
  const float* xr = x + (size_t)bb * In;
  float s = bias[o];
  for (int j = 0; j < In; ++j) s += xr[j] * W[(size_t)j * On + o];
  if (act == 1) s = fmaxf(s, 0.f);
  else if (act == 2) s = 1.f / (1.f + __expf(-s));
  y[i] = s;
}

// batch-norm over batch axis, in place; blockDim == Bn == 128, grid == On
__global__ void k_bn(float* __restrict__ h, const float* __restrict__ g,
                     const float* __restrict__ be, int Bn, int On) {
  int o = blockIdx.x, t = threadIdx.x;
  __shared__ float s1[128], s2[128];
  float v = h[(size_t)t * On + o];
  s1[t] = v; s2[t] = v * v; __syncthreads();
  for (int st = 64; st > 0; st >>= 1) {
    if (t < st) { s1[t] += s1[t + st]; s2[t] += s2[t + st]; }
    __syncthreads();
  }
  float m = s1[0] / (float)Bn;
  float var = s2[0] / (float)Bn - m * m;
  h[(size_t)t * On + o] = (v - m) * rsqrtf(var + 1e-5f) * g[o] + be[o];
}

__global__ void k_sample(const float* __restrict__ mu, const float* __restrict__ sg,
                         const float* __restrict__ noise, float* __restrict__ zs, int n) {
  int i = blockIdx.x * blockDim.x + threadIdx.x;
  if (i < n) zs[i] = mu[i] + noise[i] * __expf(sg[i]);
}

// ---------------------------------------------------------------------------
// A_pred assembly + BCE loss reduction (writes A_pred straight into d_out+4)
// ---------------------------------------------------------------------------
__global__ void k_assemble(const float* __restrict__ Al, const float* __restrict__ Ag,
                           const float* __restrict__ As, const float* __restrict__ A,
                           float* __restrict__ Apred, float* __restrict__ accum) {
  int i = blockIdx.x * blockDim.x + threadIdx.x;  // B*N*N
  int t = threadIdx.x;
  __shared__ float red[256];
  float loss = 0.f;
  if (i < B_ * N_ * N_) {
    int b = i / (N_ * N_), rc = i % (N_ * N_);
    int ri = rc / N_, cj = rc % N_;
    int gi = ri / NLg, li = ri % NLg, gj = cj / NLg, lj = cj % NLg;
    const float* Alb = Al + (size_t)b * 256;
    const float* Agb = Ag + (size_t)b * 256;
    const float* Asb = As + (size_t)b * 256;
    float v = 0.f;
    if (ri > cj && gi != gj) v += Agb[gi * 16 + gj] * Asb[li * 16 + lj];  // M
    if (cj > ri && gj != gi) v += Agb[gj * 16 + gi] * Asb[lj * 16 + li];  // M^T
    if (gi == gj) v += Alb[li * 16 + lj];                                 // A_tmp
    Apred[i] = v;
    float pc = fminf(fmaxf(v, 1e-12f), 1.f - 1e-12f);
    float a = A[i];
    loss = -(a * __logf(pc) + (1.f - a) * log1pf(-pc));
  }
  red[t] = loss; __syncthreads();
  for (int st = 128; st > 0; st >>= 1) {
    if (t < st) red[t] += red[t + st];
    __syncthreads();
  }
  if (t == 0) atomicAdd(accum, red[0]);
}

__global__ void k_kl(const float* __restrict__ mul, const float* __restrict__ mug,
                     const float* __restrict__ sgl, const float* __restrict__ sgg,
                     float* __restrict__ accum) {
  int i = blockIdx.x * blockDim.x + threadIdx.x;  // B*128
  int t = threadIdx.x;
  __shared__ float red[256];
  float term = 0.f;
  if (i < B_ * 128) {
    int b = i >> 7, d = i & 127;
    float mu, s;
    if (d < 64) { mu = mul[b * 64 + d]; s = sgl[b * 64 + d]; }
    else        { mu = mug[b * 64 + d - 64]; s = sgg[b * 64 + d - 64]; }
    term = -0.5f * (1.f + s - mu * mu - __expf(2.f * s));
  }
  red[t] = term; __syncthreads();
  for (int st = 128; st > 0; st >>= 1) {
    if (t < st) red[t] += red[t + st];
    __syncthreads();
  }
  if (t == 0) atomicAdd(accum + 1, red[0]);
}

__global__ void k_nrm(const float* __restrict__ mu, float* __restrict__ nrm) {
  int i = blockIdx.x * blockDim.x + threadIdx.x;
  if (i >= B_) return;
  float s = 0.f;
  for (int d = 0; d < 64; ++d) { float v = mu[i * 64 + d]; s += v * v; }
  nrm[i] = s;
}

__global__ void k_sim(const float* __restrict__ mu, const float* __restrict__ nrm,
                      float* __restrict__ sim) {
  int i = blockIdx.x * blockDim.x + threadIdx.x;  // B*B
  if (i >= B_ * B_) return;
  int r = i / B_, c = i % B_;
  float d = 0.f;
  for (int j = 0; j < 64; ++j) d += mu[r * 64 + j] * mu[c * 64 + j];
  sim[i] = __expf(d / (sqrtf(nrm[r] * nrm[c]) * 0.2f));
}

__global__ void k_simcol(const float* __restrict__ sim, float* __restrict__ colsum) {
  int j = blockIdx.x, t = threadIdx.x;  // 128 threads
  __shared__ float red[128];
  red[t] = sim[(size_t)t * B_ + j]; __syncthreads();
  for (int st = 64; st > 0; st >>= 1) {
    if (t < st) red[t] += red[t + st];
    __syncthreads();
  }
  if (t == 0) colsum[j] = red[0];
}

__global__ void k_reg(const float* __restrict__ sim, const float* __restrict__ colsum,
                      float* __restrict__ accum) {
  int i = blockIdx.x * blockDim.x + threadIdx.x;  // B*B
  int t = threadIdx.x;
  __shared__ float red[256];
  float v = 0.f;
  if (i < B_ * B_) { int c = i % B_; v = __logf(sim[i] / colsum[c]); }
  red[t] = v; __syncthreads();
  for (int st = 128; st > 0; st >>= 1) {
    if (t < st) red[t] += red[t + st];
    __syncthreads();
  }
  if (t == 0) atomicAdd(accum + 2, red[0]);
}

__global__ void k_copytail(const float* __restrict__ mul, const float* __restrict__ mug,
                           float* __restrict__ out) {
  int i = blockIdx.x * blockDim.x + threadIdx.x;  // 2*8192
  const size_t base = 4 + (size_t)B_ * N_ * N_;
  if (i < 8192) out[base + i] = mul[i];
  else if (i < 16384) out[base + 8192 + (i - 8192)] = mug[i - 8192];
}

__global__ void k_final(const float* __restrict__ accum, float* __restrict__ out) {
  if (threadIdx.x == 0 && blockIdx.x == 0) {
    float adj = accum[0] / (float)(B_ * N_ * N_);
    float kl  = accum[1] / (float)(B_ * 128);
    float reg = -accum[2] / (float)(B_ * B_);
    out[0] = 1e8f * adj + 1e6f * kl + 1e7f * reg;
    out[1] = adj; out[2] = kl; out[3] = reg;
  }
}

// ---------------------------------------------------------------------------
// host launcher
// ---------------------------------------------------------------------------
extern "C" void kernel_launch(void* const* d_in, const int* in_sizes, int n_in,
                              void* d_out, int out_size, void* d_ws, size_t ws_size,
                              hipStream_t stream) {
  (void)in_sizes; (void)n_in; (void)out_size; (void)ws_size;

  // inputs in setup_inputs() dict order, params flattened in insertion order
  const float* A        = (const float*)d_in[0];
  const float* noise_l  = (const float*)d_in[1];
  const float* noise_g  = (const float*)d_in[2];
  const float* eps_l    = (const float*)d_in[3];
  const float* eps_g    = (const float*)d_in[4];
  const float* gin_l_W  = (const float*)d_in[5];
  const float* gin_l_b  = (const float*)d_in[6];
  const float* gin_g_W  = (const float*)d_in[7];
  const float* gin_g_b  = (const float*)d_in[8];
#define P(i) ((const float*)d_in[i])

  float* fout = (float*)d_out;

  // workspace carve-up (256B aligned)
  unsigned char* w = (unsigned char*)d_ws;
  size_t off = 0;
  auto carve = [&](size_t bytes) -> void* {
    void* p = w + off;
    off = (off + bytes + 255) & ~(size_t)255;
    return p;
  };
  const size_t NN = (size_t)N_ * N_;
  bf16*  Abf   = (bf16*)carve(B_ * NN * 2);
  bf16*  XT    = (bf16*)carve(B_ * NN * 2);
  bf16*  tmpbf = (bf16*)carve(B_ * NN * 2);
  bf16*  WT    = (bf16*)carve(8 * NN * 2);
  float* Xcur  = (float*)carve(B_ * NN * 4);
  float* zl    = (float*)carve((size_t)B_ * N_ * N_ * L_ * 4);
  float* zg    = (float*)carve((size_t)B_ * N_ * L_ * 4);
  float* hcl   = (float*)carve((size_t)B_ * N_ * K_ * 4);
  float* stats = (float*)carve(512 * 4);
  int*   idx   = (int*)  carve((size_t)B_ * N_ * 4);
  float* zlc   = (float*)carve((size_t)B_ * K_ * N_ * L_ * 4);
  float* h1    = (float*)carve((size_t)B_ * 64 * 4);
  float* mu_l  = (float*)carve((size_t)B_ * 64 * 4);
  float* sg_l  = (float*)carve((size_t)B_ * 64 * 4);
  float* mu_g  = (float*)carve((size_t)B_ * 64 * 4);
  float* sg_g  = (float*)carve((size_t)B_ * 64 * 4);
  float* zs_l  = (float*)carve((size_t)B_ * 64 * 4);
  float* zs_g  = (float*)carve((size_t)B_ * 64 * 4);
  float* dh    = (float*)carve((size_t)B_ * 64 * 4);
  float* Al    = (float*)carve((size_t)B_ * 256 * 4);
  float* Ag    = (float*)carve((size_t)B_ * 256 * 4);
  float* As    = (float*)carve((size_t)B_ * 256 * 4);
  float* nrm   = (float*)carve((size_t)B_ * 4);
  float* sim   = (float*)carve((size_t)B_ * B_ * 4);
  float* csum  = (float*)carve((size_t)B_ * 4);
  float* accum = (float*)carve(64 * 4);

  // ---- prologue ----
  k_zero<<<1, 64, 0, stream>>>(accum, 8);
  k_f2bf<<<(B_ * (int)NN + 255) / 256, 256, 0, stream>>>(A, Abf, B_ * (int)NN);
  k_wt<<<(4 * (int)NN + 255) / 256, 256, 0, stream>>>(gin_l_W, WT, 4);
  k_wt<<<(4 * (int)NN + 255) / 256, 256, 0, stream>>>(gin_g_W, WT + 4 * NN, 4);
  k_ident<<<(B_ * (int)NN + 255) / 256, 256, 0, stream>>>(Xcur, XT);

  // 64x128 macro-tile: 8 blocks per batch element
  const int gemmBlocks = B_ * 8;  // 1024

  // ---- local GIN stack (feats -> zl) ----
  for (int l = 0; l < L_; ++l) {
    k_gemm1<<<gemmBlocks, 256, 0, stream>>>(Abf, XT, Xcur, eps_l, l, tmpbf);
    k_gemm2<<<gemmBlocks, 256, 0, stream>>>(tmpbf, WT + (size_t)l * NN,
                                            gin_l_b + l * N_, Xcur,
                                            zl, N_ * L_, l * N_, XT);
  }

  // ---- global GIN stack (continues from mutated X) ----
  for (int l = 0; l < L_; ++l) {
    k_gemm1<<<gemmBlocks, 256, 0, stream>>>(Abf, XT, Xcur, eps_g, l, tmpbf);
    k_gemm2<<<gemmBlocks, 256, 0, stream>>>(tmpbf, WT + (size_t)(4 + l) * NN,
                                            gin_g_b + l * N_, Xcur,
                                            nullptr, 0, 0, XT);
    k_colsum<<<B_, 256, 0, stream>>>(Xcur, zg, l);
  }

  // ---- clustering ----
  k_cluster_mlp<<<(B_ * N_ * K_ + 255) / 256, 256, 0, stream>>>(zl, P(33), P(34), hcl);
  k_cl_stats<<<N_, 256, 0, stream>>>(hcl, stats);
  k_cl_softmax<<<B_ * K_, 256, 0, stream>>>(hcl, stats, P(35), P(36));
  k_argmax<<<(B_ * N_ + 255) / 256, 256, 0, stream>>>(hcl, idx);
  k_zlc<<<B_ * K_, 256, 0, stream>>>(zl, idx, zlc);

  // ---- heads ----
  const int g64 = (B_ * 64 + 255) / 256;
  // mu_l
  k_linear<<<g64, 256, 0, stream>>>(zlc, P(9),  P(10), h1, B_, K_ * N_ * L_, 64, 1);
  k_bn<<<64, 128, 0, stream>>>(h1, P(11), P(12), B_, 64);
  k_linear<<<g64, 256, 0, stream>>>(h1, P(13), P(14), mu_l, B_, 64, 64, 0);
  // sg_l (final relu)
  k_linear<<<g64, 256, 0, stream>>>(zlc, P(15), P(16), h1, B_, K_ * N_ * L_, 64, 1);
  k_bn<<<64, 128, 0, stream>>>(h1, P(17), P(18), B_, 64);
  k_linear<<<g64, 256, 0, stream>>>(h1, P(19), P(20), sg_l, B_, 64, 64, 1);
  // mu_g
  k_linear<<<g64, 256, 0, stream>>>(zg, P(21), P(22), h1, B_, N_ * L_, 64, 1);
  k_bn<<<64, 128, 0, stream>>>(h1, P(23), P(24), B_, 64);
  k_linear<<<g64, 256, 0, stream>>>(h1, P(25), P(26), mu_g, B_, 64, 64, 0);
  // sg_g (final relu)
  k_linear<<<g64, 256, 0, stream>>>(zg, P(27), P(28), h1, B_, N_ * L_, 64, 1);
  k_bn<<<64, 128, 0, stream>>>(h1, P(29), P(30), B_, 64);
  k_linear<<<g64, 256, 0, stream>>>(h1, P(31), P(32), sg_g, B_, 64, 64, 1);

  // ---- reparameterize ----
  k_sample<<<g64, 256, 0, stream>>>(mu_l, sg_l, noise_l, zs_l, B_ * 64);
  k_sample<<<g64, 256, 0, stream>>>(mu_g, sg_g, noise_g, zs_g, B_ * 64);

  // ---- decoder preds ----
  const int g256 = (B_ * 256 + 255) / 256;
  k_linear<<<g64, 256, 0, stream>>>(zs_l, P(37), P(38), dh, B_, 64, 64, 1);
  k_linear<<<g256, 256, 0, stream>>>(dh, P(39), P(40), Al, B_, 64, 256, 2);
  k_linear<<<g64, 256, 0, stream>>>(zs_g, P(41), P(42), dh, B_, 64, 64, 1);
  k_linear<<<g256, 256, 0, stream>>>(dh, P(43), P(44), Ag, B_, 64, 256, 2);
  k_linear<<<g64, 256, 0, stream>>>(zs_l, P(45), P(46), dh, B_, 64, 64, 1);
  k_linear<<<g256, 256, 0, stream>>>(dh, P(47), P(48), As, B_, 64, 256, 2);

  // ---- A_pred + losses ----
  k_assemble<<<(B_ * (int)NN) / 256, 256, 0, stream>>>(Al, Ag, As, A, fout + 4, accum);
  k_kl<<<(B_ * 128) / 256, 256, 0, stream>>>(mu_l, mu_g, sg_l, sg_g, accum);
  k_nrm<<<1, 128, 0, stream>>>(mu_l, nrm);
  k_sim<<<(B_ * B_) / 256, 256, 0, stream>>>(mu_l, nrm, sim);
  k_simcol<<<B_, 128, 0, stream>>>(sim, csum);
  k_reg<<<(B_ * B_) / 256, 256, 0, stream>>>(sim, csum, accum);

  // ---- outputs ----
  k_copytail<<<64, 256, 0, stream>>>(mu_l, mu_g, fout);
  k_final<<<1, 1, 0, stream>>>(accum, fout);
}
